// FeatureJointAttention_17772574671412
// MI455X (gfx1250) — matrix operand order repaired and checked by reading, not compiled, not run on hardware
//
#include <hip/hip_runtime.h>
#include <hip/hip_bf16.h>
#include <math.h>

typedef __attribute__((ext_vector_type(16))) __bf16 v16bf;
typedef __attribute__((ext_vector_type(8)))  float  v8f;

constexpr int B   = 32;
constexpr int N   = 15;
constexpr int T   = 256;
constexpr int H   = 128;
constexpr int NP1 = N + 1;      // 16
constexpr int D   = H + 3;      // 131
constexpr int QKP = 132;        // padded per-batch vector stride
constexpr int M_TOT = NP1 * T;  // 4096

// ---------------------------------------------------------------------------
// Kernel 1: q[b,h] = node(b,its)·Wq[h,:]  then  qk[b,d] = Σ_h q[h]·Wk[h,d]
// ---------------------------------------------------------------------------
__global__ void k_qk(const float* __restrict__ node, const float* __restrict__ cond,
                     const float* __restrict__ Wq,   const float* __restrict__ Wk,
                     const int*   __restrict__ its_p, float* __restrict__ qk)
{
    __shared__ float qs[H];
    const int b = blockIdx.x, tid = threadIdx.x;
    const int its = its_p[0];
    const float* nrow = node + ((size_t)b * T + its) * H;
    const float* crow = cond + ((size_t)b * T + its) * 3;
    if (tid < H) {
        const float* wr = Wq + (size_t)tid * D;
        float acc = 0.f;
        #pragma unroll 4
        for (int d2 = 0; d2 < H; ++d2) acc += nrow[d2] * wr[d2];
        acc += crow[0] * wr[H] + crow[1] * wr[H + 1] + crow[2] * wr[H + 2];
        qs[tid] = acc;
    }
    __syncthreads();
    if (tid < D) {
        float acc = 0.f;
        #pragma unroll 4
        for (int h = 0; h < H; ++h) acc += qs[h] * Wk[(size_t)h * D + tid];
        qk[(size_t)b * QKP + tid] = acc;
    }
}

// ---------------------------------------------------------------------------
// Kernel 2 (dominant, WMMA): S_t[b,n,t] = kv[b,n,t,:]·qk[b,:]
// A = 16x32 bf16 kv tile, B = qk broadcast into all 16 columns.
// One block per (b,n); 8 waves; each wave does 2 M-tiles x 4 k-steps.
// ---------------------------------------------------------------------------
__global__ void k_scores(const float* __restrict__ node, const float* __restrict__ neigh,
                         const float* __restrict__ cond, const float* __restrict__ qk_ws,
                         float* __restrict__ S)
{
    const int b    = blockIdx.x >> 4;
    const int n    = blockIdx.x & 15;
    const int tid  = threadIdx.x;
    const int wave = tid >> 5;
    const int lane = tid & 31;
    const int hi   = lane >> 4;     // half-wave select
    const int lo   = lane & 15;

    const float* qk = qk_ws + (size_t)b * QKP;

    // B fragments: B[k, col] = qk[dbase + k] for every col.
    // Layout: element e of lane -> K = e + (e>=8 ? 8 : 0) + 8*hi.
    v16bf bq[4];
    #pragma unroll
    for (int kk = 0; kk < 4; ++kk) {
        const int base = kk * 32 + hi * 8;
        #pragma unroll
        for (int e = 0; e < 16; ++e) bq[kk][e] = (__bf16)qk[base + e + (e >= 8 ? 8 : 0)];
    }
    const float s0 = qk[128], s1 = qk[129], s2 = qk[130];

    const float* src = (n == 0) ? (node + (size_t)b * T * H)
                                : (neigh + ((size_t)b * N + (n - 1)) * T * H);
    const float* cb  = cond + (size_t)b * T * 3;
    float*       Srow = S + ((size_t)b * NP1 + n) * T;

    #pragma unroll
    for (int i = 0; i < 2; ++i) {
        const int t0 = (wave * 2 + i) * 16;
        const float* p = src + (size_t)(t0 + lo) * H;
        v8f acc = {};
        #pragma unroll
        for (int kk = 0; kk < 4; ++kk) {
            // A layout: element e -> K = e + (e>=8 ? 8 : 0) + 8*hi
            const int d0 = kk * 32 + hi * 8;
            const float4 c0 = *(const float4*)(p + d0);
            const float4 c1 = *(const float4*)(p + d0 + 4);
            const float4 c2 = *(const float4*)(p + d0 + 16);
            const float4 c3 = *(const float4*)(p + d0 + 20);
            v16bf a;
            a[0]  = (__bf16)c0.x; a[1]  = (__bf16)c0.y; a[2]  = (__bf16)c0.z; a[3]  = (__bf16)c0.w;
            a[4]  = (__bf16)c1.x; a[5]  = (__bf16)c1.y; a[6]  = (__bf16)c1.z; a[7]  = (__bf16)c1.w;
            a[8]  = (__bf16)c2.x; a[9]  = (__bf16)c2.y; a[10] = (__bf16)c2.z; a[11] = (__bf16)c2.w;
            a[12] = (__bf16)c3.x; a[13] = (__bf16)c3.y; a[14] = (__bf16)c3.z; a[15] = (__bf16)c3.w;
            acc = __builtin_amdgcn_wmma_f32_16x16x32_bf16(
                      false, a, false, bq[kk], (short)0, acc, false, false);
        }
        // D[m, col] identical across columns; col 0 lanes publish.
        if (lo == 0) {
            #pragma unroll
            for (int r = 0; r < 8; ++r) {
                const int t = t0 + hi * 8 + r;     // C/D layout: VGPR r -> M = r + 8*hi
                const float* cr = cb + (size_t)t * 3;
                Srow[t] = acc[r] + cr[0] * s0 + cr[1] * s1 + cr[2] * s2;
            }
        }
    }
}

// ---------------------------------------------------------------------------
// Kernel 3: per-batch S-norm + masked softmax over 4096 scores.
// Writes A (d_out + B*H) and S_norm (d_out + B*H + B*NP1*T).
// ---------------------------------------------------------------------------
__global__ void k_softmax(const float* __restrict__ S, const unsigned char* __restrict__ mask,
                          const int* __restrict__ its_p, float* __restrict__ out)
{
    __shared__ float red[T];
    const int b = blockIdx.x, tid = threadIdx.x;   // tid == k index
    const int its = its_p[0];
    const unsigned char mk = mask[((size_t)b * T + its) * T + tid];

    float vals[NP1];
    float sq = 0.f, lmax = -INFINITY;
    #pragma unroll
    for (int nn = 0; nn < NP1; ++nn) {
        const float v = S[((size_t)b * NP1 + nn) * T + tid];
        sq += v * v;                                // norm uses pre-mask S
        const float mv = mk ? -INFINITY : v;
        vals[nn] = mv;
        lmax = fmaxf(lmax, mv);
    }
    red[tid] = sq; __syncthreads();
    for (int s = T / 2; s > 0; s >>= 1) { if (tid < s) red[tid] += red[tid + s]; __syncthreads(); }
    const float norm = sqrtf(red[0]); __syncthreads();

    red[tid] = lmax; __syncthreads();
    for (int s = T / 2; s > 0; s >>= 1) { if (tid < s) red[tid] = fmaxf(red[tid], red[tid + s]); __syncthreads(); }
    const float gmax = red[0]; __syncthreads();

    float e[NP1];
    float esum = 0.f;
    #pragma unroll
    for (int nn = 0; nn < NP1; ++nn) { e[nn] = __expf(vals[nn] - gmax); esum += e[nn]; }
    red[tid] = esum; __syncthreads();
    for (int s = T / 2; s > 0; s >>= 1) { if (tid < s) red[tid] += red[tid + s]; __syncthreads(); }
    const float inv = 1.f / red[0];

    float* A = out + B * H;
    #pragma unroll
    for (int nn = 0; nn < NP1; ++nn)
        A[((size_t)b * NP1 + nn) * T + tid] = e[nn] * inv;
    if (tid == 0) out[B * H + (size_t)B * NP1 * T + b] = norm;
}

// ---------------------------------------------------------------------------
// Kernel 4: z[b,d] = Σ_m A[b,m]·kv[b,m,d]  (second pass over kv, L2-resident).
// 8 segment-blocks per batch, atomicAdd into zeroed z.
// ---------------------------------------------------------------------------
__global__ void k_z(const float* __restrict__ node, const float* __restrict__ neigh,
                    const float* __restrict__ cond, const float* __restrict__ out,
                    float* __restrict__ z)
{
    const int b   = blockIdx.x >> 3;
    const int seg = blockIdx.x & 7;
    const int d   = threadIdx.x;                    // 0..127
    const float* A  = out + B * H + (size_t)b * M_TOT;
    const float* cb = cond + (size_t)b * T * 3;

    float acc = 0.f;
    float acc_c = 0.f;                              // only meaningful for d<3
    #pragma unroll
    for (int j = 0; j < 2; ++j) {
        const int nn = seg * 2 + j;
        const float* src = (nn == 0) ? node + (size_t)b * T * H
                                     : neigh + ((size_t)b * N + (nn - 1)) * T * H;
        const float* An = A + (size_t)nn * T;
        for (int k = 0; k < T; ++k) {
            const float a = An[k];
            acc += a * src[(size_t)k * H + d];
            if (d < 3) acc_c += a * cb[(size_t)k * 3 + d];
        }
    }
    atomicAdd(&z[(size_t)b * QKP + d], acc);
    if (d < 3) atomicAdd(&z[(size_t)b * QKP + H + d], acc_c);
}

// ---------------------------------------------------------------------------
// Kernel 5: output[b,h] = Σ_d z[b,d]·Wv[h,d]
// ---------------------------------------------------------------------------
__global__ void k_out(const float* __restrict__ z, const float* __restrict__ Wv,
                      float* __restrict__ out)
{
    const int b = blockIdx.x, h = threadIdx.x;
    const float* zb = z + (size_t)b * QKP;
    const float* w  = Wv + (size_t)h * D;
    float acc = 0.f;
    #pragma unroll 4
    for (int d = 0; d < D; ++d) acc += zb[d] * w[d];
    out[(size_t)b * H + h] = acc;
}

// ---------------------------------------------------------------------------
extern "C" void kernel_launch(void* const* d_in, const int* in_sizes, int n_in,
                              void* d_out, int out_size, void* d_ws, size_t ws_size,
                              hipStream_t stream)
{
    (void)in_sizes; (void)n_in; (void)out_size; (void)ws_size;
    const float* node  = (const float*)d_in[0];   // [B,T,H]
    const float* neigh = (const float*)d_in[1];   // [B,N,T,H]
    const float* cond  = (const float*)d_in[2];   // [B,T,3]
    /* d_in[3] = neighbors_number: unused by the reference */
    const int*   its   = (const int*)d_in[4];     // scalar (T-1)
    const unsigned char* mask = (const unsigned char*)d_in[5]; // [B,T,T] bool
    const float* Wq    = (const float*)d_in[6];   // [H, H+3]
    const float* Wk    = (const float*)d_in[7];
    const float* Wv    = (const float*)d_in[8];
    float* out = (float*)d_out;

    // workspace layout (floats): qk | z | S_t   (~0.55 MB total)
    float* ws = (float*)d_ws;
    float* qk = ws;
    float* z  = ws + (size_t)B * QKP;
    float* S  = ws + (size_t)2 * B * QKP;

    hipMemsetAsync(z, 0, (size_t)B * QKP * sizeof(float), stream);
    k_qk     <<<B,        160, 0, stream>>>(node, cond, Wq, Wk, its, qk);
    k_scores <<<B * NP1,  256, 0, stream>>>(node, neigh, cond, qk, S);
    k_softmax<<<B,        T,   0, stream>>>(S, mask, its, out);
    k_z      <<<B * 8,    H,   0, stream>>>(node, neigh, cond, out, z);
    k_out    <<<B,        H,   0, stream>>>(z, Wv, out);
}
